// DualGCNNet_69724499083527
// MI455X (gfx1250) — compile-verified, matrix operand order checked by reading, not run on hardware
//
#include <hip/hip_runtime.h>

#define NEQ   10000
#define NGEN  2048
#define BATCH 1024
#define HIDN  2048
#define MU    4096
#define LAMB  4096
#define NEDGE 30000
#define NF    (NEQ + NGEN)   // 12048
#define NOUT  (MU + LAMB)    // 8192

typedef __attribute__((ext_vector_type(16))) __bf16 bf16x16;
typedef __attribute__((ext_vector_type(8)))  __bf16 bf16x8;
typedef __attribute__((ext_vector_type(8)))  float  f32x8;

// ---------------- bf16 split helpers (a ~= hi + lo, both bf16, RNE) ----------------
static __device__ __forceinline__ void split_native(float a, __bf16& h, __bf16& l) {
    h = (__bf16)a;                 // RNE f32->bf16
    l = (__bf16)(a - (float)h);    // residual, RNE
}
static __device__ __forceinline__ unsigned pack_split(float a) {
    __bf16 h, l; split_native(a, h, l);
    return ((unsigned)__builtin_bit_cast(unsigned short, h) << 16)
         |  (unsigned)__builtin_bit_cast(unsigned short, l);
}

// ---------------- GCN stage (replicates reference exactly, incl. flat-index quirk) ----
__global__ void k_gcn_init(float* __restrict__ deg, float* __restrict__ agg) {
    int i = blockIdx.x * blockDim.x + threadIdx.x;
    if (i < NEQ) { deg[i] = 1.0f; agg[i] = 0.0f; }
}

__global__ void k_gcn_deg(const long long* __restrict__ ei, float* __restrict__ deg) {
    int e = blockIdx.x * blockDim.x + threadIdx.x;
    if (e < NEDGE) {
        int dst = (int)ei[NEDGE + e];
        atomicAdd(&deg[dst], (float)BATCH);
    }
}

__global__ void k_gcn_msg(const long long* __restrict__ ei,
                          const float* __restrict__ x,
                          const float* __restrict__ gw,
                          const float* __restrict__ deg,
                          float* __restrict__ agg) {
    int e = blockIdx.x * blockDim.x + threadIdx.x;
    if (e < NEDGE) {
        int src = (int)ei[e];
        int dst = (int)ei[NEDGE + e];
        // x_flat[k] = xd[k % B][k / B] = x[(k%B)*NF + (k/B)]   (k = src < NEQ)
        float xs  = x[(size_t)(src % BATCH) * NF + (src / BATCH)];
        float msg = (float)BATCH * rsqrtf(deg[src]) * rsqrtf(deg[dst]) * (gw[0] * xs);
        atomicAdd(&agg[dst], msg);
    }
}

// writes feat as packed (bf16_hi<<16)|bf16_lo per element
__global__ void k_build_feat(const float* __restrict__ x,
                             const float* __restrict__ gw,
                             const float* __restrict__ gb,
                             const float* __restrict__ deg,
                             const float* __restrict__ agg,
                             unsigned* __restrict__ feat) {
    int i = blockIdx.x * blockDim.x + threadIdx.x;   // over BATCH*NF
    if (i >= BATCH * NF) return;
    int b = i / NF;
    int n = i - b * NF;
    float xv = x[i];
    float v;
    if (n < NEQ) {
        int   k = n * BATCH + b;                 // flat index into (NEQ*B) array
        float h = gw[0] * xv;
        if (k < NEQ) v = agg[k] + h / deg[k];    // only flat idx < NEQ got edge agg
        else         v = h;                      // deg==1, agg==0
        v = fmaxf(v + gb[0], 0.0f);
    } else {
        v = xv;                                  // generator features pass through
    }
    feat[i] = pack_split(v);
}

// ---------------- bf16x3 split WMMA GEMM ----------------
// C(M,N) = [relu](A(M,K) @ B(K,N) + bias)
// A: packed (hi|lo) u32 per element, row-major. B: f32, split in-kernel.
// Block tile 128x128, 8 waves (2x4), wave tile 64x32 = 4x2 WMMA 16x16 tiles.
#define LDT 40   // LDS row stride in bf16 elems (80B: keeps b128 frag loads 16B-aligned)

static __device__ __forceinline__ bf16x16 frag16(const __bf16* p, int kb) {
    bf16x8 a = *(const bf16x8*)(p + kb);        // K = kb..kb+7
    bf16x8 b = *(const bf16x8*)(p + 16 + kb);   // K = 16+kb..16+kb+7
    bf16x16 f;
#pragma unroll
    for (int i = 0; i < 8; ++i) { f[i] = a[i]; f[i + 8] = b[i]; }
    return f;
}

__global__ __launch_bounds__(256) void gemm_bf16x3(
    const unsigned* __restrict__ Apk, const float* __restrict__ Bm,
    const float* __restrict__ bias, void* __restrict__ Cout,
    int M, int N, int K, int do_relu, int pack_out, int split_out)
{
    __shared__ __align__(16) __bf16 sAh[128][LDT];
    __shared__ __align__(16) __bf16 sAl[128][LDT];
    __shared__ __align__(16) __bf16 sBh[128][LDT];   // B stored column-major per tile
    __shared__ __align__(16) __bf16 sBl[128][LDT];

    const int tid  = threadIdx.x;
    const int lane = tid & 31;
    const int wave = tid >> 5;
    const int wm   = wave >> 2;       // 0..1  -> 64 rows each
    const int wn   = wave & 3;        // 0..3  -> 32 cols each
    const int bm   = blockIdx.y * 128;
    const int bn   = blockIdx.x * 128;

    f32x8 acc[4][2] = {};

    for (int kt = 0; kt < K; kt += 32) {
        const bool kfull = (kt + 32 <= K);   // uniform; false at most once (K%32==16)

        // ---- stage A tile (packed u32 -> hi/lo LDS), rows 0..127, K kt..kt+31 ----
        {
            const int r = tid >> 1;           // 0..127
            const int h = tid & 1;            // which 16-wide K half
            const unsigned* ap = Apk + (size_t)(bm + r) * K + kt + h * 16;
            if (kfull) {
#pragma unroll
                for (int g = 0; g < 2; ++g) {
                    bf16x8 hv, lv;
#pragma unroll
                    for (int i = 0; i < 8; ++i) {
                        unsigned w = ap[g * 8 + i];
                        hv[i] = __builtin_bit_cast(__bf16, (unsigned short)(w >> 16));
                        lv[i] = __builtin_bit_cast(__bf16, (unsigned short)(w & 0xFFFFu));
                    }
                    *(bf16x8*)&sAh[r][h * 16 + g * 8] = hv;
                    *(bf16x8*)&sAl[r][h * 16 + g * 8] = lv;
                }
            } else {
#pragma unroll
                for (int g = 0; g < 2; ++g) {
                    bf16x8 hv, lv;
#pragma unroll
                    for (int i = 0; i < 8; ++i) {
                        int kg = kt + h * 16 + g * 8 + i;
                        unsigned w = (kg < K) ? ap[g * 8 + i] : 0u;
                        hv[i] = __builtin_bit_cast(__bf16, (unsigned short)(w >> 16));
                        lv[i] = __builtin_bit_cast(__bf16, (unsigned short)(w & 0xFFFFu));
                    }
                    *(bf16x8*)&sAh[r][h * 16 + g * 8] = hv;
                    *(bf16x8*)&sAl[r][h * 16 + g * 8] = lv;
                }
            }
        }
        // ---- stage B tile (f32 -> split, transposed to [col][k]) ----
        {
            const int c  = tid & 127;         // local col
            const int kh = tid >> 7;          // 0..1
            const float* bp = Bm + (size_t)(kt + kh * 16) * N + bn + c;
            if (kt + 32 < K)                   // prefetch next k-tile of weights
                __builtin_prefetch(bp + (size_t)32 * N, 0, 1);
            if (kfull) {
#pragma unroll
                for (int g = 0; g < 2; ++g) {
                    bf16x8 hv, lv;
#pragma unroll
                    for (int i = 0; i < 8; ++i) {
                        float v = bp[(size_t)(g * 8 + i) * N];
                        __bf16 hh, ll; split_native(v, hh, ll);
                        hv[i] = hh; lv[i] = ll;
                    }
                    *(bf16x8*)&sBh[c][kh * 16 + g * 8] = hv;
                    *(bf16x8*)&sBl[c][kh * 16 + g * 8] = lv;
                }
            } else {
#pragma unroll
                for (int g = 0; g < 2; ++g) {
                    bf16x8 hv, lv;
#pragma unroll
                    for (int i = 0; i < 8; ++i) {
                        int kg = kt + kh * 16 + g * 8 + i;
                        float v = (kg < K) ? bp[(size_t)(g * 8 + i) * N] : 0.0f;
                        __bf16 hh, ll; split_native(v, hh, ll);
                        hv[i] = hh; lv[i] = ll;
                    }
                    *(bf16x8*)&sBh[c][kh * 16 + g * 8] = hv;
                    *(bf16x8*)&sBl[c][kh * 16 + g * 8] = lv;
                }
            }
        }
        __syncthreads();

        // ---- fragments: ISA 16-bit A layout, lane l: row = l&15,
        //      elems 0..7 -> K = 8*(l>>4)+0..7, elems 8..15 -> K = 16+8*(l>>4)+0..7
        const int fr = lane & 15;
        const int kb = (lane >> 4) << 3;

        bf16x16 ah[4], al[4];
#pragma unroll
        for (int mi = 0; mi < 4; ++mi) {
            const int r = wm * 64 + mi * 16 + fr;
            ah[mi] = frag16(&sAh[r][0], kb);
            al[mi] = frag16(&sAl[r][0], kb);
        }
        bf16x16 bh[2], bl[2];
#pragma unroll
        for (int ni = 0; ni < 2; ++ni) {
            const int c = wn * 32 + ni * 16 + fr;
            bh[ni] = frag16(&sBh[c][0], kb);
            bl[ni] = frag16(&sBl[c][0], kb);
        }

#pragma unroll
        for (int mi = 0; mi < 4; ++mi)
#pragma unroll
            for (int ni = 0; ni < 2; ++ni) {
                acc[mi][ni] = __builtin_amdgcn_wmma_f32_16x16x32_bf16(
                    false, ah[mi], false, bh[ni], (short)0, acc[mi][ni], false, false);
                acc[mi][ni] = __builtin_amdgcn_wmma_f32_16x16x32_bf16(
                    false, ah[mi], false, bl[ni], (short)0, acc[mi][ni], false, false);
                acc[mi][ni] = __builtin_amdgcn_wmma_f32_16x16x32_bf16(
                    false, al[mi], false, bh[ni], (short)0, acc[mi][ni], false, false);
            }
        __syncthreads();
    }

    // ---- epilogue: C/D layout lane l: col = l&15; VGPR v: row = v + 8*(l>>4)
    const int cn = lane & 15;
    const int rb = (lane >> 4) * 8;
#pragma unroll
    for (int mi = 0; mi < 4; ++mi) {
#pragma unroll
        for (int ni = 0; ni < 2; ++ni) {
            const int col = bn + wn * 32 + ni * 16 + cn;
            const float bv = bias[col];
#pragma unroll
            for (int v = 0; v < 8; ++v) {
                const int row = bm + wm * 64 + mi * 16 + rb + v;
                float val = acc[mi][ni][v] + bv;
                if (do_relu) val = fmaxf(val, 0.0f);
                if (pack_out) {
                    ((unsigned*)Cout)[(size_t)row * N + col] = pack_split(val);
                } else if (split_out) {
                    // tuple-flat: mu block (cols <4096) then lamb block
                    size_t idx = (size_t)(col >> 12) * ((size_t)M * 4096)
                               + (size_t)row * 4096 + (size_t)(col & 4095);
                    ((float*)Cout)[idx] = val;
                } else {
                    ((float*)Cout)[(size_t)row * N + col] = val;
                }
            }
        }
    }
}

// ---------------- launch ----------------
extern "C" void kernel_launch(void* const* d_in, const int* in_sizes, int n_in,
                              void* d_out, int out_size, void* d_ws, size_t ws_size,
                              hipStream_t stream) {
    const float*     x    = (const float*)d_in[0];
    const long long* ei   = (const long long*)d_in[1];   // int64 (2, E)
    const float*     gw   = (const float*)d_in[2];
    const float*     gb   = (const float*)d_in[3];
    const float*     W1   = (const float*)d_in[4];
    const float*     b1   = (const float*)d_in[5];
    const float*     Wout = (const float*)d_in[6];
    const float*     bout = (const float*)d_in[7];
    float*           out  = (float*)d_out;

    float*    deg  = (float*)d_ws;                       // NEQ f32
    float*    agg  = deg + NEQ;                          // NEQ f32
    unsigned* feat = (unsigned*)(agg + NEQ);             // BATCH*NF packed
    unsigned* hid  = feat + (size_t)BATCH * NF;          // BATCH*HIDN packed

    k_gcn_init<<<(NEQ   + 255) / 256, 256, 0, stream>>>(deg, agg);
    k_gcn_deg <<<(NEDGE + 255) / 256, 256, 0, stream>>>(ei, deg);
    k_gcn_msg <<<(NEDGE + 255) / 256, 256, 0, stream>>>(ei, x, gw, deg, agg);
    k_build_feat<<<((size_t)BATCH * NF + 255) / 256, 256, 0, stream>>>(
        x, gw, gb, deg, agg, feat);

    dim3 g1(HIDN / 128, BATCH / 128);   // 16 x 8
    gemm_bf16x3<<<g1, 256, 0, stream>>>(feat, W1, b1, (void*)hid,
                                        BATCH, HIDN, NF, /*relu*/1, /*pack*/1, /*split*/0);
    dim3 g2(NOUT / 128, BATCH / 128);   // 64 x 8
    gemm_bf16x3<<<g2, 256, 0, stream>>>(hid, Wout, bout, (void*)out,
                                        BATCH, NOUT, HIDN, /*relu*/0, /*pack*/0, /*split*/1);
}
